// Qwen3MoeAttention_32933809226592
// MI455X (gfx1250) — matrix operand order repaired and checked by reading, not compiled, not run on hardware
//
#include <hip/hip_runtime.h>
#include <hip/hip_bf16.h>
#include <cstdint>

// ---------------- Problem constants ----------------
#define S_LEN   2048
#define HDIM    2048
#define NHEADS  32
#define NKVH    4
#define GRP     8
#define DHEAD   128
#define SCALE_F 0.08838834764831845f   // 1/sqrt(128)
#define NEG_F   (-1e30f)

// ---------------- Vector types ----------------
typedef __attribute__((ext_vector_type(16))) __bf16 v16bf;
typedef __attribute__((ext_vector_type(8)))  __bf16 v8bf;
typedef __attribute__((ext_vector_type(8)))  float  v8f;
typedef __attribute__((ext_vector_type(4)))  float  v4f;
typedef __attribute__((ext_vector_type(4)))  int    v4i;
typedef __attribute__((address_space(3)))    v4i    as3_v4i;  // LDS-qualified int4

// ---------------- CDNA5 async copy (global -> LDS, ASYNCcnt) ----------------
#if defined(__gfx1250__) && __has_builtin(__builtin_amdgcn_global_load_async_to_lds_b128)
#define HAVE_ASYNC 1
#else
#define HAVE_ASYNC 0
#endif

static __device__ __forceinline__ void async_cp16(const __bf16* g, __bf16* l) {
#if HAVE_ASYNC
    __builtin_amdgcn_global_load_async_to_lds_b128((v4i*)g, (as3_v4i*)l, 0, 0);
#else
    *(v8bf*)l = *(const v8bf*)g;
#endif
}

static __device__ __forceinline__ void wait_async0() {
#if HAVE_ASYNC
#if __has_builtin(__builtin_amdgcn_s_wait_asynccnt)
    __builtin_amdgcn_s_wait_asynccnt(0);
#else
    asm volatile("s_wait_asynccnt 0" ::: "memory");
#endif
#endif
}

static __device__ __forceinline__ __bf16 f2bf(float f) {
    unsigned u = __builtin_bit_cast(unsigned, f);
    u += 0x7FFFu + ((u >> 16) & 1u);          // round-to-nearest-even
    unsigned short h = (unsigned short)(u >> 16);
    return __builtin_bit_cast(__bf16, h);
}

// A-matrix fragment (16-bit, 16x32): low lanes hold K=0..15, high lanes K=16..31.
static __device__ __forceinline__ v16bf frag_split(const __bf16* p) {
    v8bf a = *(const v8bf*)p;
    v8bf b = *(const v8bf*)(p + 8);
    v16bf r;
#pragma unroll
    for (int i = 0; i < 8; ++i) { r[i] = a[i]; r[i + 8] = b[i]; }
    return r;
}

// B-matrix fragment (16-bit, 32x16): each lane holds 16 contiguous K values.
static __device__ __forceinline__ v16bf frag_contig(const __bf16* p) {
    v8bf a = *(const v8bf*)p;
    v8bf b = *(const v8bf*)(p + 8);
    v16bf r;
#pragma unroll
    for (int i = 0; i < 8; ++i) { r[i] = a[i]; r[i + 8] = b[i]; }
    return r;
}

static __device__ __forceinline__ v8f vzero8() {
    v8f z;
#pragma unroll
    for (int i = 0; i < 8; ++i) z[i] = 0.0f;
    return z;
}

static __device__ __forceinline__ v8f wmma_bf16(v16bf a, v16bf b, v8f c) {
    return __builtin_amdgcn_wmma_f32_16x16x32_bf16(false, a, false, b,
                                                   (short)0, c, false, false);
}

// =====================================================================
// Kernel 0a: elementwise fp32 -> bf16 (hidden states), 4 elems/thread
// =====================================================================
__global__ __launch_bounds__(256)
void cvt_bf16_kernel(const float* __restrict__ X, __bf16* __restrict__ Y, int n4) {
    int i = (blockIdx.x * 256 + threadIdx.x);
    if (i < n4) {
        v4f x = *(const v4f*)(X + (size_t)i * 4);
        __bf16* y = Y + (size_t)i * 4;
#pragma unroll
        for (int j = 0; j < 4; ++j) y[j] = f2bf(x[j]);
    }
}

// =====================================================================
// Kernel 0b: B[K][N] fp32 -> Bt[N][K] bf16 (weight transpose+convert)
// =====================================================================
__global__ __launch_bounds__(256)
void transpose_bf16_kernel(const float* __restrict__ B, __bf16* __restrict__ Bt,
                           int K, int N) {
    __shared__ float tile[32][33];
    const int tx = threadIdx.x & 31;
    const int ty = threadIdx.x >> 5;          // 0..7
    const int kb = blockIdx.x * 32;
    const int nb = blockIdx.y * 32;
#pragma unroll
    for (int i = 0; i < 4; ++i)
        tile[ty + i * 8][tx] = B[(size_t)(kb + ty + i * 8) * N + nb + tx];
    __syncthreads();
#pragma unroll
    for (int i = 0; i < 4; ++i)
        Bt[(size_t)(nb + ty + i * 8) * K + kb + tx] = f2bf(tile[tx][ty + i * 8]);
}

// =====================================================================
// Kernel 1: C[M,N] = A[M,K] @ Bt[N,K]^T, both bf16, fp32 out.
// 256 thr (8 waves), tile 128x128, K-step 32, double-buffered async LDS.
// Wave grid 2x4: each wave 64 rows x 32 cols (4x2 WMMA frags).
// =====================================================================
#define LDSA_STRIDE 40   // bf16 elems, padded (80B) -> conflict-free b128 reads
__global__ __launch_bounds__(256)
void gemm_wmma_kernel(const __bf16* __restrict__ A,
                      const __bf16* __restrict__ Bt,
                      float* __restrict__ C,
                      int M, int N, int K) {
    __shared__ __attribute__((aligned(16))) __bf16 lA[2][128 * LDSA_STRIDE];
    __shared__ __attribute__((aligned(16))) __bf16 lB[2][128 * LDSA_STRIDE];

    const int t    = threadIdx.x;
    const int lane = t & 31;
    const int wave = t >> 5;
    const int wrow = wave >> 2;   // 0..1 -> 64 rows
    const int wcol = wave & 3;    // 0..3 -> 32 cols
    const int r    = lane & 15;
    const int hi   = lane >> 4;
    const int mb   = blockIdx.y;
    const int nb   = blockIdx.x;

    v8f acc[4][2];
#pragma unroll
    for (int mt = 0; mt < 4; ++mt)
#pragma unroll
        for (int nt = 0; nt < 2; ++nt) acc[mt][nt] = vzero8();

    // staging assignment: thread -> (row-in-tile, k-half)
    const int sm = t >> 1;        // 0..127
    const int sh = t & 1;         // 0..1
    const __bf16* Abase = A  + (size_t)(mb * 128 + sm) * K + sh * 16;
    const __bf16* Bbase = Bt + (size_t)(nb * 128 + sm) * K + sh * 16;
    const int ldsoff = sm * LDSA_STRIDE + sh * 16;

    auto stage = [&](int buf, int kt) {
        async_cp16(Abase + kt,     &lA[buf][ldsoff]);
        async_cp16(Abase + kt + 8, &lA[buf][ldsoff + 8]);
        async_cp16(Bbase + kt,     &lB[buf][ldsoff]);
        async_cp16(Bbase + kt + 8, &lB[buf][ldsoff + 8]);
    };

    const int nk = K >> 5;
    stage(0, 0);
    for (int it = 0; it < nk; ++it) {
        const int cur = it & 1;
        wait_async0();            // my copies into buf cur are done
        __syncthreads();          // everyone's copies done; prev reads of buf cur^1 done
        if (it + 1 < nk) stage(cur ^ 1, (it + 1) * 32);   // DMA overlaps compute

        v16bf af[4], bfr[2];
#pragma unroll
        for (int mt = 0; mt < 4; ++mt) {
            int row = wrow * 64 + mt * 16 + r;
            af[mt] = frag_split(&lA[cur][row * LDSA_STRIDE + hi * 16]);
        }
#pragma unroll
        for (int nt = 0; nt < 2; ++nt) {
            int col = wcol * 32 + nt * 16 + r;
            bfr[nt] = frag_contig(&lB[cur][col * LDSA_STRIDE + hi * 16]);
        }
#pragma unroll
        for (int mt = 0; mt < 4; ++mt)
#pragma unroll
            for (int nt = 0; nt < 2; ++nt)
                acc[mt][nt] = wmma_bf16(af[mt], bfr[nt], acc[mt][nt]);
    }

#pragma unroll
    for (int mt = 0; mt < 4; ++mt)
#pragma unroll
        for (int nt = 0; nt < 2; ++nt) {
            int col = nb * 128 + wcol * 32 + nt * 16 + r;
#pragma unroll
            for (int i = 0; i < 8; ++i) {
                int row = mb * 128 + wrow * 64 + mt * 16 + i + hi * 8;
                C[(size_t)row * N + col] = acc[mt][nt][i];
            }
        }
}

// =====================================================================
// Kernel 2: per-head RMS norm + RoPE (Q,K) / convert (V) -> bf16
// grid (S, 40): y<32 => Q head, 32..35 => K head, 36..39 => V head
// =====================================================================
__global__ __launch_bounds__(128)
void norm_rope_kernel(const float* __restrict__ Qraw,
                      const float* __restrict__ Kraw,
                      const float* __restrict__ Vraw,
                      const float* __restrict__ qw,
                      const float* __restrict__ kw,
                      const float* __restrict__ cosb,
                      const float* __restrict__ sinb,
                      __bf16* __restrict__ Qh,
                      __bf16* __restrict__ Kh,
                      __bf16* __restrict__ Vh) {
    const int s  = blockIdx.x;
    const int hh = blockIdx.y;
    const int d  = threadIdx.x;

    const float* src; const float* w = nullptr; __bf16* dst; bool rope;
    if (hh < NHEADS) {
        src = Qraw + (size_t)s * (NHEADS * DHEAD) + hh * DHEAD;
        w = qw; dst = Qh + ((size_t)hh * S_LEN + s) * DHEAD; rope = true;
    } else if (hh < NHEADS + NKVH) {
        int kv = hh - NHEADS;
        src = Kraw + (size_t)s * (NKVH * DHEAD) + kv * DHEAD;
        w = kw; dst = Kh + ((size_t)kv * S_LEN + s) * DHEAD; rope = true;
    } else {
        int kv = hh - NHEADS - NKVH;
        src = Vraw + (size_t)s * (NKVH * DHEAD) + kv * DHEAD;
        dst = Vh + ((size_t)kv * S_LEN + s) * DHEAD; rope = false;
    }

    float x = src[d];
    float out;
    if (rope) {
        float ss = x * x;
#pragma unroll
        for (int off = 1; off < 32; off <<= 1) ss += __shfl_xor(ss, off, 32);
        __shared__ float part[4];
        if ((d & 31) == 0) part[d >> 5] = ss;
        __syncthreads();
        float tot = part[0] + part[1] + part[2] + part[3];
        float inv = rsqrtf(tot * (1.0f / 128.0f) + 1e-6f);
        float y = x * inv * w[d];
        int pd = d ^ 64;                       // rotate-half partner
        float rot = ((d < 64) ? -1.0f : 1.0f) * src[pd] * inv * w[pd];
        out = y * cosb[(size_t)s * DHEAD + d] + rot * sinb[(size_t)s * DHEAD + d];
    } else {
        out = x;
    }
    dst[d] = f2bf(out);
}

// =====================================================================
// Kernel 3: flash attention, causal, GQA. Block = 1 head x 64 q rows,
// 4 waves, each wave owns 16 q rows end-to-end. KV tile = 64 keys.
// K tile staged via async-to-LDS; V staged transposed for P@V.
// =====================================================================
__global__ __launch_bounds__(128)
void attn_kernel(const __bf16* __restrict__ Qh,
                 const __bf16* __restrict__ Kh,
                 const __bf16* __restrict__ Vh,
                 __bf16* __restrict__ AttnOut) {
    __shared__ __attribute__((aligned(16))) __bf16 lK[64 * 136];   // [key][d]
    __shared__ __attribute__((aligned(16))) __bf16 lVt[128 * 72];  // [d][key]
    __shared__ __attribute__((aligned(16))) __bf16 lP[4][16 * 72]; // per-wave P

    const int qb   = blockIdx.x;        // q block of 64
    const int h    = blockIdx.y;
    const int kv   = h >> 3;            // GQA: h = kv*GRP + g
    const int t    = threadIdx.x;
    const int lane = t & 31;
    const int wave = t >> 5;
    const int r    = lane & 15;
    const int hi   = lane >> 4;
    const int qr0  = qb * 64 + wave * 16;

    // Q fragments straight from global (A-layout), kept live all kernel
    v16bf qf[4];
    {
        const __bf16* qp = Qh + ((size_t)h * S_LEN + qr0 + r) * DHEAD;
#pragma unroll
        for (int c = 0; c < 4; ++c)
            qf[c] = frag_split(qp + c * 32 + hi * 16);
    }

    v8f oacc[8];
#pragma unroll
    for (int i = 0; i < 8; ++i) oacc[i] = vzero8();
    float m_run[8], l_run[8];
#pragma unroll
    for (int i = 0; i < 8; ++i) { m_run[i] = NEG_F; l_run[i] = 0.0f; }

    const __bf16* Kbase = Kh + (size_t)kv * S_LEN * DHEAD;
    const __bf16* Vbase = Vh + (size_t)kv * S_LEN * DHEAD;

    for (int kb = 0; kb <= qb; ++kb) {
        // ---- stage K (async, row major) and V (transposed) tiles ----
        {
            const int key  = t >> 1;
            const int half = t & 1;
            const __bf16* ksrc = Kbase + (size_t)(kb * 64 + key) * DHEAD + half * 64;
            __bf16* kdst = &lK[key * 136 + half * 64];
#pragma unroll
            for (int i = 0; i < 8; ++i)
                async_cp16(ksrc + i * 8, kdst + i * 8);
            const __bf16* vsrc = Vbase + (size_t)(kb * 64 + key) * DHEAD + half * 64;
#pragma unroll
            for (int i = 0; i < 8; ++i) {
                v8bf vv = *(const v8bf*)(vsrc + i * 8);
#pragma unroll
                for (int j = 0; j < 8; ++j)
                    lVt[(half * 64 + i * 8 + j) * 72 + key] = vv[j];
            }
        }
        wait_async0();
        __syncthreads();

        // ---- S = Q @ K^T (16 q rows x 64 keys per wave) ----
        v8f sacc[4];
#pragma unroll
        for (int nt = 0; nt < 4; ++nt) sacc[nt] = vzero8();
#pragma unroll
        for (int nt = 0; nt < 4; ++nt)
#pragma unroll
            for (int c = 0; c < 4; ++c) {
                v16bf bfr = frag_contig(&lK[(nt * 16 + r) * 136 + c * 32 + hi * 16]);
                sacc[nt] = wmma_bf16(qf[c], bfr, sacc[nt]);
            }

        // ---- scale + causal mask ----
        const bool diag = (kb == qb);
#pragma unroll
        for (int nt = 0; nt < 4; ++nt)
#pragma unroll
            for (int i = 0; i < 8; ++i) {
                float v = sacc[nt][i] * SCALE_F;
                if (diag) {
                    int qrow = qr0 + i + hi * 8;
                    int kcol = kb * 64 + nt * 16 + r;
                    if (kcol > qrow) v = NEG_F;
                }
                sacc[nt][i] = v;
            }

        // ---- online softmax (row reductions across 16-lane halves) ----
#pragma unroll
        for (int i = 0; i < 8; ++i) {
            float mx = fmaxf(fmaxf(sacc[0][i], sacc[1][i]),
                             fmaxf(sacc[2][i], sacc[3][i]));
#pragma unroll
            for (int off = 1; off < 16; off <<= 1)
                mx = fmaxf(mx, __shfl_xor(mx, off, 32));
            float mnew  = fmaxf(m_run[i], mx);
            float alpha = __expf(m_run[i] - mnew);
            float ps = 0.0f;
#pragma unroll
            for (int nt = 0; nt < 4; ++nt) {
                float p = __expf(sacc[nt][i] - mnew);
                sacc[nt][i] = p;
                ps += p;
            }
#pragma unroll
            for (int off = 1; off < 16; off <<= 1)
                ps += __shfl_xor(ps, off, 32);
            l_run[i] = l_run[i] * alpha + ps;
            m_run[i] = mnew;
#pragma unroll
            for (int nt2 = 0; nt2 < 8; ++nt2) oacc[nt2][i] *= alpha;
        }

        // ---- P -> per-wave LDS (C-layout -> A-layout re-stripe) ----
        __bf16* Pw = &lP[wave][0];
#pragma unroll
        for (int nt = 0; nt < 4; ++nt)
#pragma unroll
            for (int i = 0; i < 8; ++i)
                Pw[(i + hi * 8) * 72 + nt * 16 + r] = f2bf(sacc[nt][i]);
        asm volatile("" ::: "memory");  // same-wave DS ops are in-order (DScnt)

        // ---- O += P @ V ----
        v16bf pf[2];
#pragma unroll
        for (int kc = 0; kc < 2; ++kc)
            pf[kc] = frag_split(Pw + r * 72 + kc * 32 + hi * 16);
#pragma unroll
        for (int nt2 = 0; nt2 < 8; ++nt2)
#pragma unroll
            for (int kc = 0; kc < 2; ++kc) {
                v16bf bfr = frag_contig(&lVt[(nt2 * 16 + r) * 72 + kc * 32 + hi * 16]);
                oacc[nt2] = wmma_bf16(pf[kc], bfr, oacc[nt2]);
            }
        __syncthreads();  // all waves done with lK/lVt before next tile
    }

    // ---- epilogue: O / l, store bf16 [S][NH*D] ----
#pragma unroll
    for (int i = 0; i < 8; ++i) {
        float recip = 1.0f / l_run[i];
        int q = qr0 + i + hi * 8;
#pragma unroll
        for (int nt2 = 0; nt2 < 8; ++nt2)
            AttnOut[(size_t)q * (NHEADS * DHEAD) + h * DHEAD + nt2 * 16 + r] =
                f2bf(oacc[nt2][i] * recip);
    }
}

// =====================================================================
// Launch
// =====================================================================
extern "C" void kernel_launch(void* const* d_in, const int* in_sizes, int n_in,
                              void* d_out, int out_size, void* d_ws, size_t ws_size,
                              hipStream_t stream) {
    const float* hidden = (const float*)d_in[0];
    const float* wq     = (const float*)d_in[1];
    const float* wk     = (const float*)d_in[2];
    const float* wv     = (const float*)d_in[3];
    const float* wo     = (const float*)d_in[4];
    const float* qnw    = (const float*)d_in[5];
    const float* knw    = (const float*)d_in[6];
    const float* cosb   = (const float*)d_in[7];
    const float* sinb   = (const float*)d_in[8];
    float* out = (float*)d_out;

    char* ws = (char*)d_ws;
    __bf16* hidB = (__bf16*)ws; ws += (size_t)S_LEN * HDIM * 2;
    __bf16* wqT  = (__bf16*)ws; ws += (size_t)(NHEADS * DHEAD) * HDIM * 2;
    __bf16* wkT  = (__bf16*)ws; ws += (size_t)(NKVH * DHEAD) * HDIM * 2;
    __bf16* wvT  = (__bf16*)ws; ws += (size_t)(NKVH * DHEAD) * HDIM * 2;
    __bf16* woT  = (__bf16*)ws; ws += (size_t)HDIM * (NHEADS * DHEAD) * 2;
    float*  Qraw = (float*)ws;  ws += (size_t)S_LEN * NHEADS * DHEAD * 4;
    float*  Kraw = (float*)ws;  ws += (size_t)S_LEN * NKVH * DHEAD * 4;
    float*  Vraw = (float*)ws;  ws += (size_t)S_LEN * NKVH * DHEAD * 4;
    __bf16* Qh   = (__bf16*)ws; ws += (size_t)NHEADS * S_LEN * DHEAD * 2;
    __bf16* Kh   = (__bf16*)ws; ws += (size_t)NKVH * S_LEN * DHEAD * 2;
    __bf16* Vh   = (__bf16*)ws; ws += (size_t)NKVH * S_LEN * DHEAD * 2;
    __bf16* Ao   = (__bf16*)ws;

    dim3 blk256(256);
    // One-time precision/layout prep (bandwidth-trivial, L2-resident)
    cvt_bf16_kernel<<<dim3((S_LEN * HDIM / 4 + 255) / 256), blk256, 0, stream>>>(
        hidden, hidB, S_LEN * HDIM / 4);
    transpose_bf16_kernel<<<dim3(HDIM / 32, (NHEADS * DHEAD) / 32), blk256, 0, stream>>>(
        wq, wqT, HDIM, NHEADS * DHEAD);
    transpose_bf16_kernel<<<dim3(HDIM / 32, (NKVH * DHEAD) / 32), blk256, 0, stream>>>(
        wk, wkT, HDIM, NKVH * DHEAD);
    transpose_bf16_kernel<<<dim3(HDIM / 32, (NKVH * DHEAD) / 32), blk256, 0, stream>>>(
        wv, wvT, HDIM, NKVH * DHEAD);
    transpose_bf16_kernel<<<dim3((NHEADS * DHEAD) / 32, HDIM / 32), blk256, 0, stream>>>(
        wo, woT, NHEADS * DHEAD, HDIM);

    // QKV projections (pure bf16 WMMA, async double-buffered)
    gemm_wmma_kernel<<<dim3(32, 16), blk256, 0, stream>>>(
        hidB, wqT, Qraw, S_LEN, NHEADS * DHEAD, HDIM);
    gemm_wmma_kernel<<<dim3(4, 16), blk256, 0, stream>>>(
        hidB, wkT, Kraw, S_LEN, NKVH * DHEAD, HDIM);
    gemm_wmma_kernel<<<dim3(4, 16), blk256, 0, stream>>>(
        hidB, wvT, Vraw, S_LEN, NKVH * DHEAD, HDIM);
    // RMS norm + RoPE + bf16 pack
    norm_rope_kernel<<<dim3(S_LEN, NHEADS + 2 * NKVH), dim3(128), 0, stream>>>(
        Qraw, Kraw, Vraw, qnw, knw, cosb, sinb, Qh, Kh, Vh);
    // Flash attention
    attn_kernel<<<dim3(S_LEN / 64, NHEADS), dim3(128), 0, stream>>>(Qh, Kh, Vh, Ao);
    // Output projection
    gemm_wmma_kernel<<<dim3(16, 16), blk256, 0, stream>>>(
        Ao, woT, out, S_LEN, HDIM, NHEADS * DHEAD);
}
